// GCBlock_25099788877921
// MI455X (gfx1250) — matrix operand order repaired and checked by reading, not compiled, not run on hardware
//
#include <hip/hip_runtime.h>

typedef __attribute__((ext_vector_type(16))) __bf16          v16bf;
typedef __attribute__((ext_vector_type(16))) unsigned short  v16u;
typedef __attribute__((ext_vector_type(8)))  unsigned short  v8u;
typedef __attribute__((ext_vector_type(8)))  float           v8f;
typedef unsigned short u16;

#define HDIM 256
#define FDIM 128
#define RDIM 32
#define CATK 384
#define CAT_STRIDE 392   // bf16 elems, padded (784B rows)
#define GEL_STRIDE 136   // bf16 elems, padded (272B rows)
#define MST_STRIDE 132   // f32 elems, padded (528B rows)

static __device__ __forceinline__ u16 f2bf_u(float f) {
  unsigned u = __float_as_uint(f);
  u += 0x7FFFu + ((u >> 16) & 1u);      // round-to-nearest-even
  return (u16)(u >> 16);
}

static __device__ __forceinline__ v8f vzero8() {
  v8f z;
#pragma unroll
  for (int t = 0; t < 8; ++t) z[t] = 0.0f;
  return z;
}

// A tile 16x32 (MxK) from global fp32 row-major matrix, converted to bf16.
// Lane L<16: row=L, K = k0+0..7 (VGPR0-3) and k0+16..23 (VGPR4-7)
// Lane L>=16: row=L-16, K = k0+8..15 and k0+24..31      (ISA 7.12.2)
static __device__ __forceinline__ v16bf loadA_gf32(const float* base, int ld,
                                                   int row0, int rmax, int k0, int lane) {
  int r = row0 + (lane & 15);
  if (r > rmax) r = rmax;
  int kb = k0 + ((lane >> 4) << 3);
  const float* p = base + (size_t)r * ld + kb;
  v16u av;
#pragma unroll
  for (int t = 0; t < 8; ++t) { av[t] = f2bf_u(p[t]); av[t + 8] = f2bf_u(p[t + 16]); }
  return __builtin_bit_cast(v16bf, av);
}

// A tile 16x32 from LDS bf16 (row-major, padded stride in elements)
static __device__ __forceinline__ v16bf loadA_lds(const u16* base, int stride,
                                                  int k0, int lane) {
  int r = lane & 15;
  int kb = k0 + ((lane >> 4) << 3);
  const u16* p = base + r * stride + kb;
  v8u lo = *(const v8u*)p;
  v8u hi = *(const v8u*)(p + 16);
  v16u av;
#pragma unroll
  for (int t = 0; t < 8; ++t) { av[t] = lo[t]; av[t + 8] = hi[t]; }
  return __builtin_bit_cast(v16bf, av);
}

// B tile 32x16 (KxN) for output cols ncol0..+15 from bf16 weight W[out][K] row-major.
// b[h] = W[ncol0 + lane%16][k0 + 8*(lane/16) + h] (h<8), +16 for h>=8  -> B[k][n] = W[n][k]
static __device__ __forceinline__ v16bf loadB(const u16* W, int K,
                                              int ncol0, int k0, int lane) {
  const u16* p = W + (size_t)(ncol0 + (lane & 15)) * K + k0 + ((lane >> 4) << 3);
  v8u lo = *(const v8u*)p;
  v8u hi = *(const v8u*)(p + 16);
  v16u bv;
#pragma unroll
  for (int t = 0; t < 8; ++t) { bv[t] = lo[t]; bv[t + 8] = hi[t]; }
  return __builtin_bit_cast(v16bf, bv);
}

#define WMMA_BF16(a, b, c) \
  __builtin_amdgcn_wmma_f32_16x16x32_bf16(false, (a), false, (b), (short)0, (c), false, false)

// ---------------------------------------------------------------------------
// Kernel 0a: fp32 -> bf16 weight conversion
__global__ void cvt_bf16_kernel(const float* __restrict__ s, u16* __restrict__ d, int n) {
  int i = blockIdx.x * 256 + threadIdx.x;
  if (i < n) d[i] = f2bf_u(s[i]);
}

// Kernel 0b: zero the fp32 aggregation buffer
__global__ void zero_kernel(float4* __restrict__ p, int n4) {
  int i = blockIdx.x * 256 + threadIdx.x;
  if (i < n4) p[i] = make_float4(0.f, 0.f, 0.f, 0.f);
}

// ---------------------------------------------------------------------------
// Kernel 1: xf = layernorm(x @ ntm_w^T + ntm_b) -> bf16 [N][128]
__global__ __launch_bounds__(32) void node_kernel(
    const float* __restrict__ x, const u16* __restrict__ ntm_wb,
    const float* __restrict__ ntm_b, const float* __restrict__ norm_g,
    const float* __restrict__ norm_b, u16* __restrict__ xfb, int N) {
  __shared__ __align__(16) u16 stage[16 * GEL_STRIDE];
  int lane = threadIdx.x;
  int lo = lane & 15, hi = lane >> 4;
  int row0 = blockIdx.x * 16;

  v8f acc[8];
#pragma unroll
  for (int nt = 0; nt < 8; ++nt) acc[nt] = vzero8();

  for (int k0 = 0; k0 < HDIM; k0 += 32) {
    v16bf a = loadA_gf32(x, HDIM, row0, N - 1, k0, lane);
#pragma unroll
    for (int nt = 0; nt < 8; ++nt) {
      v16bf b = loadB(ntm_wb, HDIM, nt * 16, k0, lane);
      acc[nt] = WMMA_BF16(a, b, acc[nt]);
    }
  }
  // bias (col = nt*16 + lo is constant across the 8 rows held per accumulator)
#pragma unroll
  for (int nt = 0; nt < 8; ++nt) {
    float bb = ntm_b[nt * 16 + lo];
#pragma unroll
    for (int v = 0; v < 8; ++v) acc[nt][v] += bb;
  }
  // LayerNorm stats per row (row = v + 8*hi); xor masks <16 stay in the half-wave
  float mean[8], rstd[8];
#pragma unroll
  for (int v = 0; v < 8; ++v) {
    float s1 = 0.f, s2 = 0.f;
#pragma unroll
    for (int nt = 0; nt < 8; ++nt) { float t = acc[nt][v]; s1 += t; s2 += t * t; }
#pragma unroll
    for (int m = 1; m < 16; m <<= 1) { s1 += __shfl_xor(s1, m, 32); s2 += __shfl_xor(s2, m, 32); }
    float mu = s1 * (1.0f / 128.0f);
    float var = s2 * (1.0f / 128.0f) - mu * mu;
    mean[v] = mu;
    rstd[v] = rsqrtf(var + 1e-5f);
  }
#pragma unroll
  for (int nt = 0; nt < 8; ++nt) {
    int col = nt * 16 + lo;
    float g = norm_g[col], be = norm_b[col];
#pragma unroll
    for (int v = 0; v < 8; ++v) {
      float val = (acc[nt][v] - mean[v]) * rstd[v] * g + be;
      stage[(v + 8 * hi) * GEL_STRIDE + col] = f2bf_u(val);
    }
  }
  __syncthreads();
  for (int r = 0; r < 16; ++r) {
    if (row0 + r < N)
      *(uint2*)(xfb + (size_t)(row0 + r) * FDIM + lane * 4) =
          *(const uint2*)(stage + r * GEL_STRIDE + lane * 4);
  }
}

// ---------------------------------------------------------------------------
// Kernel 2: fused per-edge pipeline. One wave per 16-edge tile.
__global__ __launch_bounds__(32) void edge_kernel(
    const float* __restrict__ rbf, const int* __restrict__ ei, const int* __restrict__ ej,
    const u16* __restrict__ xfb,
    const u16* __restrict__ rtm_wb, const float* __restrict__ rtm_b,
    const u16* __restrict__ l1_wb, const float* __restrict__ l1_b,
    const float* __restrict__ ln_g, const float* __restrict__ ln_b,
    const u16* __restrict__ l2_wb, const float* __restrict__ l2_b,
    float* __restrict__ agg, int E) {
  __shared__ __align__(16) unsigned char smem[12800];
  __shared__ int il[16], jl[16];
  u16* cat = (u16*)smem;                  // phase A: cat bf16 [16][CAT_STRIDE]
  u16* gel = (u16*)smem;                  // phase B: gelu(h) bf16 [16][GEL_STRIDE]
  float* mst = (float*)(smem + 4352);     // phase B: m fp32 [16][MST_STRIDE]

  int lane = threadIdx.x;
  int lo = lane & 15, hi = lane >> 4;
  int e0 = blockIdx.x * 16;

  if (lane < 16) {
    int e = e0 + lane; if (e >= E) e = E - 1;
    il[lane] = ei[e]; jl[lane] = ej[e];
  }
  __syncthreads();

  // Async gather of xf[i], xf[j] rows straight into LDS (ASYNCcnt path,
  // no VGPR round-trip). Each lane moves 8B; 32 lanes cover one 256B row.
  {
    unsigned cat_base = (unsigned)(uintptr_t)cat;   // LDS byte address (addr[31:0])
    for (int r = 0; r < 16; ++r) {
      int ii = il[r], jj = jl[r];
      const u16* srcI = xfb + (size_t)ii * FDIM + lane * 4;
      const u16* srcJ = xfb + (size_t)jj * FDIM + lane * 4;
      unsigned dstI = cat_base + (unsigned)(r * CAT_STRIDE + lane * 4) * 2u;
      unsigned dstJ = dstI + 256u;                  // +128 bf16 elems
      asm volatile("global_load_async_to_lds_b64 %0, %1, off"
                   :: "v"(dstI), "v"(srcI) : "memory");
      asm volatile("global_load_async_to_lds_b64 %0, %1, off"
                   :: "v"(dstJ), "v"(srcJ) : "memory");
    }
  }

  // Meanwhile: edge_attr = rbf @ rtm_w^T + rtm_b -> cat cols 256..383 (K=32)
  {
    int erow = e0 + lo; if (erow >= E) erow = E - 1;
    const float* p = rbf + (size_t)erow * RDIM + (hi << 3);
    v16u av;
#pragma unroll
    for (int t = 0; t < 8; ++t) { av[t] = f2bf_u(p[t]); av[t + 8] = f2bf_u(p[t + 16]); }
    v16bf a = __builtin_bit_cast(v16bf, av);
#pragma unroll
    for (int nt = 0; nt < 8; ++nt) {
      v16bf b = loadB(rtm_wb, RDIM, nt * 16, 0, lane);
      v8f c = vzero8();
      c = WMMA_BF16(a, b, c);
      float bb = rtm_b[nt * 16 + lo];
#pragma unroll
      for (int v = 0; v < 8; ++v)
        cat[(v + 8 * hi) * CAT_STRIDE + 256 + nt * 16 + lo] = f2bf_u(c[v] + bb);
    }
  }
  asm volatile("s_wait_asynccnt 0x0" ::: "memory");   // gathers landed in LDS
  __syncthreads();

  // GEMM1: h = cat[16,384] @ l1_w^T   (12 K-steps x 8 N-tiles = 96 WMMA)
  v8f acc[8];
#pragma unroll
  for (int nt = 0; nt < 8; ++nt) acc[nt] = vzero8();
  for (int k0 = 0; k0 < CATK; k0 += 32) {
    v16bf a = loadA_lds(cat, CAT_STRIDE, k0, lane);
#pragma unroll
    for (int nt = 0; nt < 8; ++nt) {
      v16bf b = loadB(l1_wb, CATK, nt * 16, k0, lane);
      acc[nt] = WMMA_BF16(a, b, acc[nt]);
    }
  }
#pragma unroll
  for (int nt = 0; nt < 8; ++nt) {
    float bb = l1_b[nt * 16 + lo];
#pragma unroll
    for (int v = 0; v < 8; ++v) acc[nt][v] += bb;
  }

  // LayerNorm (shuffle reduce) + exact GELU -> gel bf16 (reuses cat space)
  float mean[8], rstd[8];
#pragma unroll
  for (int v = 0; v < 8; ++v) {
    float s1 = 0.f, s2 = 0.f;
#pragma unroll
    for (int nt = 0; nt < 8; ++nt) { float t = acc[nt][v]; s1 += t; s2 += t * t; }
#pragma unroll
    for (int m = 1; m < 16; m <<= 1) { s1 += __shfl_xor(s1, m, 32); s2 += __shfl_xor(s2, m, 32); }
    float mu = s1 * (1.0f / 128.0f);
    float var = s2 * (1.0f / 128.0f) - mu * mu;
    mean[v] = mu;
    rstd[v] = rsqrtf(var + 1e-5f);
  }
#pragma unroll
  for (int nt = 0; nt < 8; ++nt) {
    int col = nt * 16 + lo;
    float g = ln_g[col], be = ln_b[col];
#pragma unroll
    for (int v = 0; v < 8; ++v) {
      float xh = (acc[nt][v] - mean[v]) * rstd[v] * g + be;
      float ge = 0.5f * xh * (1.0f + erff(xh * 0.70710678118f));
      gel[(v + 8 * hi) * GEL_STRIDE + col] = f2bf_u(ge);
    }
  }
  __syncthreads();

  // GEMM2: m = gelu(h)[16,128] @ l2_w^T  (4 K-steps x 8 N-tiles = 32 WMMA)
  v8f acc2[8];
#pragma unroll
  for (int nt = 0; nt < 8; ++nt) acc2[nt] = vzero8();
  for (int k0 = 0; k0 < FDIM; k0 += 32) {
    v16bf a = loadA_lds(gel, GEL_STRIDE, k0, lane);
#pragma unroll
    for (int nt = 0; nt < 8; ++nt) {
      v16bf b = loadB(l2_wb, FDIM, nt * 16, k0, lane);
      acc2[nt] = WMMA_BF16(a, b, acc2[nt]);
    }
  }
  // bias + stage m in LDS for coalesced scatter
#pragma unroll
  for (int nt = 0; nt < 8; ++nt) {
    float bb = l2_b[nt * 16 + lo];
#pragma unroll
    for (int v = 0; v < 8; ++v)
      mst[(v + 8 * hi) * MST_STRIDE + nt * 16 + lo] = acc2[nt][v] + bb;
  }
  __syncthreads();

  // segment-sum scatter: coalesced 128B atomic rows into L2-resident agg
  for (int r = 0; r < 16; ++r) {
    if (e0 + r < E) {
      int jj = jl[r];
      float* dst = agg + (size_t)jj * FDIM + lane;
      const float* src = mst + r * MST_STRIDE + lane;
#pragma unroll
      for (int q = 0; q < 4; ++q) atomicAdd(dst + 32 * q, src[32 * q]);
    }
  }
}

// ---------------------------------------------------------------------------
// Kernel 3: out = agg @ mtn_w^T  ([N,128] x [128,256]); one wave per 16x128 tile
__global__ __launch_bounds__(32) void out_kernel(
    const float* __restrict__ agg, const u16* __restrict__ mtn_wb,
    float* __restrict__ out, int N) {
  __shared__ __align__(16) float mo[16 * MST_STRIDE];
  int lane = threadIdx.x;
  int lo = lane & 15, hi = lane >> 4;
  int bid = blockIdx.x;
  int half = bid & 1;
  int row0 = (bid >> 1) * 16;

  v8f acc[8];
#pragma unroll
  for (int nt = 0; nt < 8; ++nt) acc[nt] = vzero8();
  for (int k0 = 0; k0 < FDIM; k0 += 32) {
    v16bf a = loadA_gf32(agg, FDIM, row0, N - 1, k0, lane);
#pragma unroll
    for (int nt = 0; nt < 8; ++nt) {
      v16bf b = loadB(mtn_wb, FDIM, half * 128 + nt * 16, k0, lane);
      acc[nt] = WMMA_BF16(a, b, acc[nt]);
    }
  }
#pragma unroll
  for (int nt = 0; nt < 8; ++nt)
#pragma unroll
    for (int v = 0; v < 8; ++v)
      mo[(v + 8 * hi) * MST_STRIDE + nt * 16 + lo] = acc[nt][v];
  __syncthreads();
  for (int r = 0; r < 16; ++r) {
    if (row0 + r < N)
      *(float4*)(out + (size_t)(row0 + r) * HDIM + half * FDIM + lane * 4) =
          *(const float4*)(mo + r * MST_STRIDE + lane * 4);
  }
}

// ---------------------------------------------------------------------------
extern "C" void kernel_launch(void* const* d_in, const int* in_sizes, int n_in,
                              void* d_out, int out_size, void* d_ws, size_t ws_size,
                              hipStream_t stream) {
  const float* x     = (const float*)d_in[0];
  const float* rbf   = (const float*)d_in[1];
  const int*   eidx  = (const int*)d_in[2];
  const float* ntm_w = (const float*)d_in[3];
  const float* ntm_b = (const float*)d_in[4];
  const float* rtm_w = (const float*)d_in[5];
  const float* rtm_b = (const float*)d_in[6];
  const float* ng    = (const float*)d_in[7];
  const float* nb_   = (const float*)d_in[8];
  const float* l1_w  = (const float*)d_in[9];
  const float* l1_b  = (const float*)d_in[10];
  const float* ln_g  = (const float*)d_in[11];
  const float* ln_b  = (const float*)d_in[12];
  const float* l2_w  = (const float*)d_in[13];
  const float* l2_b  = (const float*)d_in[14];
  const float* mtn_w = (const float*)d_in[15];

  int N = in_sizes[0] / HDIM;   // 50000
  int E = in_sizes[1] / RDIM;   // 800000
  const int* ei = eidx;
  const int* ej = eidx + E;

  unsigned char* ws = (unsigned char*)d_ws;
  u16* ntm_wb = (u16*)(ws + 0);        // 128*256 bf16
  u16* rtm_wb = (u16*)(ws + 65536);    // 128*32
  u16* l1_wb  = (u16*)(ws + 73728);    // 128*384
  u16* l2_wb  = (u16*)(ws + 172032);   // 128*128
  u16* mtn_wb = (u16*)(ws + 204800);   // 256*128
  u16* xfb    = (u16*)(ws + 270336);   // N*128 bf16
  float* agg  = (float*)(ws + 270336 + (size_t)N * FDIM * 2);  // N*128 fp32

  cvt_bf16_kernel<<<(128 * 256 + 255) / 256, 256, 0, stream>>>(ntm_w, ntm_wb, 128 * 256);
  cvt_bf16_kernel<<<(128 * 32 + 255) / 256, 256, 0, stream>>>(rtm_w, rtm_wb, 128 * 32);
  cvt_bf16_kernel<<<(128 * 384 + 255) / 256, 256, 0, stream>>>(l1_w, l1_wb, 128 * 384);
  cvt_bf16_kernel<<<(128 * 128 + 255) / 256, 256, 0, stream>>>(l2_w, l2_wb, 128 * 128);
  cvt_bf16_kernel<<<(256 * 128 + 255) / 256, 256, 0, stream>>>(mtn_w, mtn_wb, 256 * 128);

  int n4 = (N * FDIM) / 4;
  zero_kernel<<<(n4 + 255) / 256, 256, 0, stream>>>((float4*)agg, n4);

  node_kernel<<<(N + 15) / 16, 32, 0, stream>>>(x, ntm_wb, ntm_b, ng, nb_, xfb, N);
  edge_kernel<<<(E + 15) / 16, 32, 0, stream>>>(rbf, ei, ej, xfb, rtm_wb, rtm_b,
                                                l1_wb, l1_b, ln_g, ln_b, l2_wb, l2_b,
                                                agg, E);
  out_kernel<<<((N + 15) / 16) * 2, 32, 0, stream>>>(agg, mtn_wb, (float*)d_out, N);
}
